// Transformer_28467043238464
// MI455X (gfx1250) — matrix operand order repaired and checked by reading, not compiled, not run on hardware
//
#include <hip/hip_runtime.h>
#include <hip/hip_bf16.h>
#include <cmath>

// ---------------- problem constants ----------------
constexpr int CB   = 4;      // batch
constexpr int CN   = 1024;   // seq len
constexpr int CDIM = 768;
constexpr int CH   = 12;     // heads
constexpr int CDH  = 64;     // head dim
constexpr int CMLP = 3072;
constexpr int CD   = 6;      // layers
constexpr int CM   = CB * CN;          // 4096 tokens
constexpr int C3HD = 3 * CH * CDH;     // 2304
constexpr float CSCALE = 0.125f;       // 64^-0.5

typedef __attribute__((ext_vector_type(16))) _Float16 v16h;
typedef __attribute__((ext_vector_type(8)))  _Float16 v8h;
typedef __attribute__((ext_vector_type(8)))  float    v8f;

__device__ inline v8f wmma32(v16h a, v16h b, v8f c) {
    return __builtin_amdgcn_wmma_f32_16x16x32_f16(
        false, a, false, b, (short)0, c, false, false);
}

__device__ inline v16h concat8(v8h lo, v8h hi) {
    return __builtin_shufflevector(lo, hi, 0, 1, 2, 3, 4, 5, 6, 7,
                                          8, 9, 10, 11, 12, 13, 14, 15);
}

// A-fragment (16x32 f16). Lane (half_, m) holds two contiguous 8-half runs:
//   elements 0..7  -> k = 8*half_ + 0..7
//   elements 8..15 -> k = 16 + 8*half_ + 0..7
// `row` points at LDS row of matrix row m (k contiguous), 16B-aligned chunks.
__device__ inline v16h a_frag(const _Float16* row, int half_) {
    v8h lo = *(const v8h*)(row + 8 * half_);
    v8h hi = *(const v8h*)(row + 16 + 8 * half_);
    return concat8(lo, hi);
}

// B-fragment (32x16 f16). Lane (half_, n) holds two contiguous 8-half runs:
//   elements 0..7  -> k = 8*half_ + 0..7
//   elements 8..15 -> k = 16 + 8*half_ + 0..7
// `row` points at LDS row of *transposed* B (i.e. [n][k], k contiguous).
__device__ inline v16h b_frag(const _Float16* row, int half_) {
    v8h lo = *(const v8h*)(row + 8 * half_);
    v8h hi = *(const v8h*)(row + 16 + 8 * half_);
    return concat8(lo, hi);
}

// ---------------- weight transpose + f16 convert: W[K,Nc] -> Wt[Nc,K] ----------------
__global__ __launch_bounds__(256)
void wt_kernel(const float* __restrict__ W, _Float16* __restrict__ Wt, int K, int Nc)
{
    __shared__ float T[32][33];
    const int k0 = blockIdx.y * 32, n0 = blockIdx.x * 32;
    const int c  = threadIdx.x & 31;
    const int r0 = threadIdx.x >> 5;
#pragma unroll
    for (int rr = 0; rr < 4; ++rr) {
        const int r = r0 + rr * 8;
        T[r][c] = W[(size_t)(k0 + r) * Nc + n0 + c];
    }
    __syncthreads();
#pragma unroll
    for (int rr = 0; rr < 4; ++rr) {
        const int r = r0 + rr * 8;
        Wt[(size_t)(n0 + r) * K + k0 + c] = (_Float16)T[c][r];
    }
}

// ---------------- fused WMMA GEMM ----------------
// A: [M,K] f16 row-major (activations). Wt: [Nc,K] f16 (pre-transposed weights).
// Block tile 128x64, 8 waves (4x2), wave tile 32x32, K-step 32.
// Inner loop is pure f16 b128 traffic; fragments load as ds_load_b128 pairs.
template <bool BIASED, bool RES, bool DOGELU, bool OUT32, bool OUT16>
__global__ __launch_bounds__(256)
void wmma_gemm(const _Float16* __restrict__ A, const _Float16* __restrict__ Wt,
               const float* __restrict__ bias, const float* __restrict__ resid,
               float* __restrict__ C32, _Float16* __restrict__ C16,
               int M, int K, int Nc)
{
    __shared__ _Float16 As[128][40];   // [m][k] row-major, stride 80B (16B aligned)
    __shared__ _Float16 Bt[64][40];    // [n][k] transposed-B, stride 80B

    const int tid   = threadIdx.x;
    const int lane  = tid & 31;
    const int wave  = tid >> 5;
    const int half_ = lane >> 4;
    const int l16   = lane & 15;
    const int wm    = wave >> 1;       // 0..3
    const int wn    = wave & 1;        // 0..1
    const int rowBase = blockIdx.y * 128;
    const int colBase = blockIdx.x * 64;

    const int ar = tid >> 1;           // 0..127
    const int ac = (tid & 1) << 4;     // 0 or 16
    const int bn = tid >> 2;           // 0..63
    const int bk = (tid & 3) << 3;     // 0,8,16,24

    v8f acc[2][2] = {};

    for (int k0 = 0; k0 < K; k0 += 32) {
        {   // stage A tile: 16 halfs per thread (2 x b128)
            const _Float16* src = A + (size_t)(rowBase + ar) * K + k0 + ac;
            *(v8h*)&As[ar][ac]     = *(const v8h*)(src);
            *(v8h*)&As[ar][ac + 8] = *(const v8h*)(src + 8);
        }
        {   // stage B tile (already n-major): 8 halfs per thread (1 x b128)
            const _Float16* src = Wt + (size_t)(colBase + bn) * K + k0 + bk;
            *(v8h*)&Bt[bn][bk] = *(const v8h*)(src);
        }
        if (k0 + 32 < K) {
            __builtin_prefetch(A  + (size_t)(rowBase + ar) * K + k0 + 32 + ac, 0, 3);
            __builtin_prefetch(Wt + (size_t)(colBase + bn) * K + k0 + 32 + bk, 0, 3);
        }
        __syncthreads();

        v16h afr[2], bfr[2];
#pragma unroll
        for (int mt = 0; mt < 2; ++mt)
            afr[mt] = a_frag(&As[wm * 32 + mt * 16 + l16][0], half_);
#pragma unroll
        for (int nt = 0; nt < 2; ++nt)
            bfr[nt] = b_frag(&Bt[wn * 32 + nt * 16 + l16][0], half_);
#pragma unroll
        for (int mt = 0; mt < 2; ++mt)
#pragma unroll
            for (int nt = 0; nt < 2; ++nt)
                acc[mt][nt] = wmma32(afr[mt], bfr[nt], acc[mt][nt]);
        __syncthreads();
    }

    // epilogue: bias -> gelu -> residual; C layout: lane(half_,l16), VGPR r -> row 8*half_+r
#pragma unroll
    for (int mt = 0; mt < 2; ++mt) {
#pragma unroll
        for (int nt = 0; nt < 2; ++nt) {
#pragma unroll
            for (int r = 0; r < 8; ++r) {
                const int row = rowBase + wm * 32 + mt * 16 + half_ * 8 + r;
                const int col = colBase + wn * 32 + nt * 16 + l16;
                float v = acc[mt][nt][r];
                if (BIASED) v += bias[col];
                if (DOGELU) v = 0.5f * v * (1.f + erff(v * 0.70710678118654752f));
                if (RES)    v += resid[(size_t)row * Nc + col];
                if (OUT32)  C32[(size_t)row * Nc + col] = v;
                if (OUT16)  C16[(size_t)row * Nc + col] = (_Float16)v;
            }
        }
    }
}

// ---------------- LayerNorm: f32 in, f16 out ----------------
__global__ __launch_bounds__(256)
void layernorm_kernel(const float* __restrict__ x, const float* __restrict__ g,
                      const float* __restrict__ b, _Float16* __restrict__ y)
{
    const int row = blockIdx.x;
    const int tid = threadIdx.x;
    const float* xr = x + (size_t)row * CDIM;

    float s = 0.f, s2 = 0.f;
    for (int i = tid; i < CDIM; i += 256) {
        const float v = xr[i];
        s += v; s2 += v * v;
    }
#pragma unroll
    for (int m = 1; m <= 16; m <<= 1) {
        s  += __shfl_xor(s,  m, 32);
        s2 += __shfl_xor(s2, m, 32);
    }
    __shared__ float r1[8], r2[8];
    __shared__ float mean_s, inv_s;
    const int lane = tid & 31, wave = tid >> 5;
    if (lane == 0) { r1[wave] = s; r2[wave] = s2; }
    __syncthreads();
    if (tid == 0) {
        float S = 0.f, S2 = 0.f;
#pragma unroll
        for (int w = 0; w < 8; ++w) { S += r1[w]; S2 += r2[w]; }
        const float mean = S * (1.f / CDIM);
        const float var  = S2 * (1.f / CDIM) - mean * mean;
        mean_s = mean;
        inv_s  = rsqrtf(var + 1e-5f);
    }
    __syncthreads();
    const float mean = mean_s, inv = inv_s;
    for (int i = tid; i < CDIM; i += 256)
        y[(size_t)row * CDIM + i] = (_Float16)((xr[i] - mean) * inv * g[i] + b[i]);
}

// ---------------- Flash attention (WMMA, online softmax), f16 in/out ----------------
// grid (N/64, H, B), 128 threads = 4 waves; wave owns 16 query rows.
__global__ __launch_bounds__(128)
void attn_kernel(const _Float16* __restrict__ qkv, _Float16* __restrict__ out)
{
    const int rt = blockIdx.x, h = blockIdx.y, b = blockIdx.z;
    const int tid   = threadIdx.x;
    const int lane  = tid & 31;
    const int wave  = tid >> 5;
    const int half_ = lane >> 4;
    const int l16   = lane & 15;

    __shared__ _Float16 Qs[64][72];      // [m][d]  (A source for Q@K^T)
    __shared__ _Float16 Ks[64][72];      // [j][d]  (transposed-B for Q@K^T: k=d contiguous)
    __shared__ _Float16 VT[64][72];      // [d][j]  (transposed-B for P@V:  k=j contiguous)
    __shared__ _Float16 Ps[4][16][72];   // per-wave P pane [m][j] (A source for P@V)

    const int qRow0 = rt * 64;

    // load Q tile: 64 x 64 halfs, v8h granules (512 / 128 threads = 4 each)
    for (int idx = tid; idx < 64 * 8; idx += 128) {
        const int r  = idx >> 3;
        const int c8 = (idx & 7) << 3;
        *(v8h*)&Qs[r][c8] =
            *(const v8h*)(qkv + (size_t)(b * CN + qRow0 + r) * C3HD + h * CDH + c8);
    }

    float mrow[8], lrow[8];
#pragma unroll
    for (int r = 0; r < 8; ++r) { mrow[r] = -3.0e38f; lrow[r] = 0.f; }
    v8f oacc[4] = {};

    for (int j0 = 0; j0 < CN; j0 += 64) {
        __syncthreads();  // previous K/V tile fully consumed
        for (int idx = tid; idx < 64 * 8; idx += 128) {
            const int r  = idx >> 3;
            const int c8 = (idx & 7) << 3;
            const size_t base = (size_t)(b * CN + j0 + r) * C3HD + h * CDH;
            *(v8h*)&Ks[r][c8] = *(const v8h*)(qkv + base + CH * CDH + c8);
            const v8h vv = *(const v8h*)(qkv + base + 2 * CH * CDH + c8);
#pragma unroll
            for (int i = 0; i < 8; ++i)
                VT[c8 + i][r] = vv[i];
        }
        __syncthreads();

        // S(16x64) = Q @ K^T ; DH=64 -> 2 k-steps of 32
        v8f sacc[4] = {};
#pragma unroll
        for (int s = 0; s < 2; ++s) {
            const v16h af = a_frag(&Qs[wave * 16 + l16][32 * s], half_);
#pragma unroll
            for (int nt = 0; nt < 4; ++nt) {
                const v16h bf = b_frag(&Ks[nt * 16 + l16][32 * s], half_);
                sacc[nt] = wmma32(af, bf, sacc[nt]);
            }
        }
#pragma unroll
        for (int nt = 0; nt < 4; ++nt)
#pragma unroll
            for (int r = 0; r < 8; ++r)
                sacc[nt][r] *= CSCALE;

        // online softmax; masks 1..8 reduce across the 16-lane column group
        float mnew[8], alpha[8], rsum[8];
#pragma unroll
        for (int r = 0; r < 8; ++r) {
            float mx = fmaxf(fmaxf(sacc[0][r], sacc[1][r]), fmaxf(sacc[2][r], sacc[3][r]));
#pragma unroll
            for (int m = 1; m <= 8; m <<= 1)
                mx = fmaxf(mx, __shfl_xor(mx, m, 32));
            const float mn = fmaxf(mrow[r], mx);
            mnew[r]  = mn;
            alpha[r] = __expf(mrow[r] - mn);
            mrow[r]  = mn;
            rsum[r]  = 0.f;
        }
#pragma unroll
        for (int nt = 0; nt < 4; ++nt)
#pragma unroll
            for (int r = 0; r < 8; ++r) {
                const float p = __expf(sacc[nt][r] - mnew[r]);
                sacc[nt][r] = p;
                rsum[r] += p;
            }
#pragma unroll
        for (int r = 0; r < 8; ++r) {
            float s = rsum[r];
#pragma unroll
            for (int m = 1; m <= 8; m <<= 1)
                s += __shfl_xor(s, m, 32);
            lrow[r] = lrow[r] * alpha[r] + s;
        }
#pragma unroll
        for (int dt = 0; dt < 4; ++dt)
#pragma unroll
            for (int r = 0; r < 8; ++r)
                oacc[dt][r] *= alpha[r];

        // spill P to per-wave pane (same-wave LDS is in-order; no block barrier needed)
#pragma unroll
        for (int nt = 0; nt < 4; ++nt)
#pragma unroll
            for (int r = 0; r < 8; ++r)
                Ps[wave][half_ * 8 + r][nt * 16 + l16] = (_Float16)sacc[nt][r];

        // O += P(16x64) @ V(64x64)
#pragma unroll
        for (int s = 0; s < 2; ++s) {
            const v16h af = a_frag(&Ps[wave][l16][32 * s], half_);
#pragma unroll
            for (int dt = 0; dt < 4; ++dt) {
                const v16h bf = b_frag(&VT[dt * 16 + l16][32 * s], half_);
                oacc[dt] = wmma32(af, bf, oacc[dt]);
            }
        }
    }

    // normalize and write f16 [B,N,H*DH]
#pragma unroll
    for (int r = 0; r < 8; ++r) {
        const int row = qRow0 + wave * 16 + half_ * 8 + r;
        const float inv = 1.f / lrow[r];
#pragma unroll
        for (int dt = 0; dt < 4; ++dt)
            out[(size_t)(b * CN + row) * (CH * CDH) + h * CDH + dt * 16 + l16] =
                (_Float16)(oacc[dt][r] * inv);
    }
}

// ---------------- helpers ----------------
__global__ void copy_f32(const float* __restrict__ src, float* __restrict__ dst, int n) {
    const int i = blockIdx.x * blockDim.x + threadIdx.x;
    if (i < n) dst[i] = src[i];
}

// pull k,v of the final layer out of the f32 QKV buffer into [B,H,N,DH]
__global__ void extract_kv(const float* __restrict__ qkv,
                           float* __restrict__ kout, float* __restrict__ vout) {
    const int i = blockIdx.x * blockDim.x + threadIdx.x;
    if (i >= CB * CH * CN * CDH) return;
    const int d  = i & 63;
    const int n  = (i >> 6) & 1023;
    const int hh = (i >> 16) % CH;
    const int b  = i / (CDH * CN * CH);
    const size_t src = (size_t)(b * CN + n) * C3HD + hh * CDH + d;
    kout[i] = qkv[src + CH * CDH];
    vout[i] = qkv[src + 2 * CH * CDH];
}

// ---------------- driver ----------------
extern "C" void kernel_launch(void* const* d_in, const int* in_sizes, int n_in,
                              void* d_out, int out_size, void* d_ws, size_t ws_size,
                              hipStream_t stream)
{
    (void)in_sizes; (void)n_in; (void)out_size; (void)ws_size;
    const float* x    = (const float*)d_in[0];
    const float* ln1g = (const float*)d_in[1];
    const float* ln1b = (const float*)d_in[2];
    const float* wqkv = (const float*)d_in[3];
    const float* wo   = (const float*)d_in[4];
    const float* bo   = (const float*)d_in[5];
    const float* ln2g = (const float*)d_in[6];
    const float* ln2b = (const float*)d_in[7];
    const float* w1   = (const float*)d_in[8];
    const float* b1   = (const float*)d_in[9];
    const float* w2   = (const float*)d_in[10];
    const float* b2   = (const float*)d_in[11];
    float* out = (float*)d_out;

    constexpr size_t SX   = (size_t)CM * CDIM;   // 3,145,728
    constexpr size_t SQKV = (size_t)CM * C3HD;   // 9,437,184
    constexpr size_t SMLP = (size_t)CM * CMLP;   // 12,582,912

    float*     X     = (float*)d_ws;             // residual stream (f32)
    float*     QKV   = X + SX;                   // f32 QKV (for k/v outputs)
    _Float16*  Hb16  = (_Float16*)(QKV + SQKV);  // LN out / attn out (f16)
    _Float16*  QKV16 = Hb16 + SX;                // f16 QKV (feeds attention)
    _Float16*  MB16  = QKV16 + SQKV;             // MLP hidden (f16)
    _Float16*  Wt    = MB16 + SMLP;              // transposed f16 weights (reused)

    copy_f32<<<(int)((SX + 255) / 256), 256, 0, stream>>>(x, X, (int)SX);

    for (int d = 0; d < CD; ++d) {
        layernorm_kernel<<<CM, 256, 0, stream>>>(X, ln1g + d * CDIM, ln1b + d * CDIM, Hb16);

        wt_kernel<<<dim3(C3HD / 32, CDIM / 32), 256, 0, stream>>>(
            wqkv + (size_t)d * CDIM * C3HD, Wt, CDIM, C3HD);
        wmma_gemm<false, false, false, true, true>
            <<<dim3(C3HD / 64, CM / 128), 256, 0, stream>>>(
                Hb16, Wt, nullptr, nullptr, QKV, QKV16, CM, CDIM, C3HD);

        if (d == CD - 1)
            extract_kv<<<(int)((SX + 255) / 256), 256, 0, stream>>>(QKV, out + SX, out + 2 * SX);

        attn_kernel<<<dim3(CN / 64, CH, CB), 128, 0, stream>>>(QKV16, Hb16);

        wt_kernel<<<dim3(CDIM / 32, CDIM / 32), 256, 0, stream>>>(
            wo + (size_t)d * CDIM * CDIM, Wt, CDIM, CDIM);
        wmma_gemm<true, true, false, true, false>
            <<<dim3(CDIM / 64, CM / 128), 256, 0, stream>>>(
                Hb16, Wt, bo + d * CDIM, X, X, nullptr, CM, CDIM, CDIM);

        layernorm_kernel<<<CM, 256, 0, stream>>>(X, ln2g + d * CDIM, ln2b + d * CDIM, Hb16);

        wt_kernel<<<dim3(CMLP / 32, CDIM / 32), 256, 0, stream>>>(
            w1 + (size_t)d * CDIM * CMLP, Wt, CDIM, CMLP);
        wmma_gemm<true, false, true, false, true>
            <<<dim3(CMLP / 64, CM / 128), 256, 0, stream>>>(
                Hb16, Wt, b1 + d * CMLP, nullptr, nullptr, MB16, CM, CDIM, CMLP);

        wt_kernel<<<dim3(CDIM / 32, CMLP / 32), 256, 0, stream>>>(
            w2 + (size_t)d * CMLP * CDIM, Wt, CMLP, CDIM);
        wmma_gemm<true, true, false, true, false>
            <<<dim3(CDIM / 64, CM / 128), 256, 0, stream>>>(
                MB16, Wt, b2 + d * CDIM, X, X, nullptr, CM, CMLP, CDIM);
    }

    copy_f32<<<(int)((SX + 255) / 256), 256, 0, stream>>>(X, out, (int)SX);
}
